// MultiHeadAttention_71244917506120
// MI455X (gfx1250) — matrix operand order repaired and checked by reading, not compiled, not run on hardware
//
#include <hip/hip_runtime.h>
#include <hip/hip_bf16.h>

// MHA forward for MI455X (gfx1250): bf16 WMMA (v_wmma_f32_16x16x32_bf16) for all
// four GEMMs, fp32 accumulate. The attention core (scores -> mask -> softmax ->
// attn@V) is fully fused through LDS: the 256MB attn tensor is written to HBM
// exactly once (it is a required output) and never re-read. The per-block q
// slab is staged into LDS by the Tensor Data Mover (tensor_load_to_lds).

typedef __bf16 bf16;
typedef __attribute__((ext_vector_type(4)))  __bf16 bf16x4;
typedef __attribute__((ext_vector_type(8)))  __bf16 bf16x8;
typedef __attribute__((ext_vector_type(16))) __bf16 bf16x16;
typedef __attribute__((ext_vector_type(8)))  float  f32x8;
typedef __attribute__((ext_vector_type(4)))  unsigned int u32x4;
typedef __attribute__((ext_vector_type(8)))  int i32x8;
typedef __attribute__((ext_vector_type(4)))  int i32x4;

#define S_LEN 2048
#define D_DIM 1024
#define H_CNT 16
#define DK_DIM 64
#define SROW 2052  // LDS row stride (floats): 2052 % 64 == 4 -> conflict-free frag reads

// TDM availability (arity differs between ROCm 7.2 clang-22 and clang-23+)
#if defined(__has_builtin)
#if __has_builtin(__builtin_amdgcn_tensor_load_to_lds) && \
    __has_builtin(__builtin_amdgcn_s_wait_tensorcnt)
#define USE_TDM 1
#endif
#endif
#ifndef USE_TDM
#define USE_TDM 0
#endif

// Byte offset of the q staging area inside the dynamic LDS block
#define LDS_QSTAGE_OFF ((16 * SROW + 32) * 4)

// ---- WMMA helpers -----------------------------------------------------------

__device__ __forceinline__ f32x8 wmma_bf16(bf16x16 a, bf16x16 b, f32x8 c) {
  // (neg_a, A, neg_b, B, c_mod, C, reuse_a, reuse_b)
  return __builtin_amdgcn_wmma_f32_16x16x32_bf16(false, a, false, b, (short)0, c,
                                                 false, false);
}

// A-matrix 16x32 bf16 fragment (ISA layout): lane group (lane>>4) selects
// koff = 0 or 8; frag elems 0..7 = A[r, k+koff+0..7], elems 8..15 = A[r, k+koff+16..23].
// Caller passes p = &A[r][k0 + koff].
__device__ __forceinline__ bf16x16 load_frag_a(const bf16* __restrict__ p) {
  bf16x8 lo = *(const bf16x8*)(p);
  bf16x8 hi = *(const bf16x8*)(p + 16);
  return __builtin_shufflevector(lo, hi, 0, 1, 2, 3, 4, 5, 6, 7,
                                 8, 9, 10, 11, 12, 13, 14, 15);
}

// B-matrix 32x16 bf16 fragment (same K order as A): lane holds column n = lane&15,
// elems 0..7 = K koff+0..7, elems 8..15 = K koff+16..23, koff = (lane>>4)*8.
// Caller passes p = &Bcol[n][k0 + (lane>>4)*8]  (column contiguous in memory).
__device__ __forceinline__ bf16x16 load_frag_b(const bf16* __restrict__ p) {
  bf16x8 lo = *(const bf16x8*)(p);
  bf16x8 hi = *(const bf16x8*)(p + 16);
  return __builtin_shufflevector(lo, hi, 0, 1, 2, 3, 4, 5, 6, 7,
                                 8, 9, 10, 11, 12, 13, 14, 15);
}

// ---- fp32 -> bf16 conversion ------------------------------------------------

__global__ void __launch_bounds__(256) cvt_f32_bf16(const float* __restrict__ in,
                                                    bf16* __restrict__ out, int n) {
  int i = (blockIdx.x * 256 + threadIdx.x) * 4;
  if (i + 3 < n) {
    float4 v = *(const float4*)(in + i);
    out[i + 0] = (bf16)v.x;
    out[i + 1] = (bf16)v.y;
    out[i + 2] = (bf16)v.z;
    out[i + 3] = (bf16)v.w;
  }
}

// ---- Generic GEMM: Y[m,n] = (sum_d A[m,d]*B[n,d] + bias[n]) * scale (+resid) ----
// A: MxK row-major, B: NxK row-major (i.e. Y = A @ B^T). One 16x16 tile / wave.

__global__ void __launch_bounds__(256) wmma_gemm_bias(
    const bf16* __restrict__ A, const bf16* __restrict__ B,
    const float* __restrict__ bias, const float* __restrict__ resid, float scale,
    bf16* __restrict__ outb, int ob_rs, int ob_cs, float* __restrict__ outf,
    int M, int N, int Kd) {
  int lane = threadIdx.x & 31;
  int wave = threadIdx.x >> 5;
  int tiles_n = N >> 4;
  int t = blockIdx.x * 8 + wave;
  if (t >= (M >> 4) * tiles_n) return;  // wave-uniform guard (EXEC stays full)
  int tm = t / tiles_n, tn = t % tiles_n;
  int r = (tm << 4) + (lane & 15);
  int c = (tn << 4) + (lane & 15);
  int ka = (lane >> 4) * 8;   // A fragment sub-offset
  int kb = (lane >> 4) * 8;   // B fragment sub-offset
  const bf16* pa = A + (size_t)r * Kd + ka;
  const bf16* pb = B + (size_t)c * Kd + kb;
  f32x8 acc = {};
  for (int k = 0; k < Kd; k += 32) {
    acc = wmma_bf16(load_frag_a(pa + k), load_frag_b(pb + k), acc);
  }
  float bn = bias ? bias[c] : 0.0f;
  int mbase = (tm << 4) + ((lane >> 4) << 3);
#pragma unroll
  for (int j = 0; j < 8; ++j) {
    int m = mbase + j;
    float v = (acc[j] + bn) * scale;
    if (resid) v += resid[(size_t)m * N + c];
    if (outf) outf[(size_t)m * N + c] = v;
    if (outb) outb[(size_t)m * ob_rs + (size_t)c * ob_cs] = (bf16)v;
  }
}

// ---- Fused attention core: scores + mask + softmax + attn@V -----------------
// One block = one (head, 16-query-row) slab.
//  P0: TDM stages the 16x64 q slab into LDS once per block (tensor_load_to_lds).
//  P1: 16x2048 masked scores -> LDS (padded stride).
//  P2: in-LDS row softmax (16-lane shfl groups); normalized probs stored to the
//      attn output (only HBM touch); per-row max & 1/sum parked in LDS.
//  P3: each wave takes a 256-wide K slice and accumulates a partial 16x64 ctx
//      tile, rebuilding bf16 prob fragments from LDS; V^T streamed from L2.
//  P4: cross-wave reduce in LDS, store ctx2 (fp32) + bf16 copy.

__global__ void __launch_bounds__(256) wmma_attn_fused(
    const bf16* __restrict__ q, const bf16* __restrict__ kmat,
    const unsigned char* __restrict__ mask, const bf16* __restrict__ vt,
    float* __restrict__ attn, float* __restrict__ ctxf, bf16* __restrict__ ctxb) {
  extern __shared__ float sc[];           // [16][SROW] + 32 stats + q stage
  float* mxa = sc + 16 * SROW;            // per-row max
  float* inva = mxa + 16;                 // per-row 1/sum
  int lane = threadIdx.x & 31;
  int wave = threadIdx.x >> 5;
  int h = blockIdx.x >> 7;                // head
  int tm = blockIdx.x & 127;              // 16-row slab index
  int ln15 = lane & 15;
  int ka = (lane >> 4) * 8;
  int kb = (lane >> 4) * 8;
  int mrow = (lane >> 4) << 3;            // 0 or 8

#if USE_TDM
  // ---- P0: DMA the 16x64 q slab into LDS with the Tensor Data Mover ----
  if (wave == 0) {
    unsigned long long va = (unsigned long long)(const void*)(
        q + (size_t)(tm << 4) * D_DIM + h * DK_DIM);
    u32x4 g0;
    g0[0] = 1u;                                  // count=1, user-mode descriptor
    g0[1] = (unsigned)LDS_QSTAGE_OFF;            // lds_addr (bytes)
    g0[2] = (unsigned)(va & 0xffffffffu);        // global_addr[31:0]
    g0[3] = (unsigned)((va >> 32) & 0x1ffffffu)  // global_addr[56:32]
            | (2u << 30);                        // type = 2 ("image")
    i32x8 g1;
    g1[0] = 0x00010000;           // workgroup_mask=0, data_size=1 (2 bytes)
    g1[1] = (int)(1024u << 16);   // tensor_dim0 = 1024 (row length)
    g1[2] = (int)(2048u << 16);   // tensor_dim1 = 2048 (rows)
    g1[3] = (int)(64u << 16);     // tile_dim0 = 64
    g1[4] = 16;                   // tile_dim1 = 16, tile_dim2 = 0
    g1[5] = 1024;                 // tensor_dim0_stride = 1024 elements
    g1[6] = 0;
    g1[7] = 0;
    i32x4 gz = {0, 0, 0, 0};
#if defined(__clang_major__) && (__clang_major__ >= 23)
    i32x8 gz8 = {0, 0, 0, 0, 0, 0, 0, 0};
    __builtin_amdgcn_tensor_load_to_lds(g0, g1, gz, gz, gz8, 0);
#else
    __builtin_amdgcn_tensor_load_to_lds(g0, g1, gz, gz, 0);
#endif
    __builtin_amdgcn_s_wait_tensorcnt(0);
  }
  __syncthreads();
  const bf16* qsrc = (const bf16*)(inva + 16);   // staged slab, rows of 64
  const bf16* pa = qsrc + ln15 * DK_DIM + ka;
#else
  const bf16* pa = q + (size_t)((tm << 4) + ln15) * D_DIM + h * DK_DIM + ka;
#endif

  // ---- P1: masked scores into LDS ----
  {
    bf16x16 a0 = load_frag_a(pa);         // q fragments reused for all 128 j-tiles
    bf16x16 a1 = load_frag_a(pa + 32);
    for (int i = 0; i < 16; ++i) {        // 8 waves x 16 iters = 128 j-tiles
      int c = ((wave + (i << 3)) << 4) + ln15;
      const bf16* pb = kmat + (size_t)c * D_DIM + h * DK_DIM + kb;
      f32x8 acc = {};
      acc = wmma_bf16(a0, load_frag_b(pb), acc);
      acc = wmma_bf16(a1, load_frag_b(pb + 32), acc);
#pragma unroll
      for (int j = 0; j < 8; ++j) {
        int m = mrow + j;
        int gm = (tm << 4) + m;
        // attn_mask True -> -1e9 (masked out), per reference jnp.where
        float v = mask[(size_t)gm * S_LEN + c] ? -1e9f : acc[j];
        sc[m * SROW + c] = v;
      }
    }
  }
  __syncthreads();

  // ---- P2: row softmax; write normalized probs to HBM (attn output) ----
  {
    int row = threadIdx.x >> 4;
    int sub = threadIdx.x & 15;
    const float* srow = sc + row * SROW;
    float mx = -3.4e38f;
    for (int i = 0; i < 32; ++i) {        // interleaved float4s: all 64 banks
      float4 x = *(const float4*)(srow + ((sub + (i << 4)) << 2));
      mx = fmaxf(mx, fmaxf(fmaxf(x.x, x.y), fmaxf(x.z, x.w)));
    }
#pragma unroll
    for (int o = 8; o > 0; o >>= 1) mx = fmaxf(mx, __shfl_xor(mx, o, 16));
    float sum = 0.0f;
    for (int i = 0; i < 32; ++i) {
      float4 x = *(const float4*)(srow + ((sub + (i << 4)) << 2));
      sum += __expf(x.x - mx) + __expf(x.y - mx) +
             __expf(x.z - mx) + __expf(x.w - mx);
    }
#pragma unroll
    for (int o = 8; o > 0; o >>= 1) sum += __shfl_xor(sum, o, 16);
    float inv = 1.0f / sum;
    if (sub == 0) {
      mxa[row] = mx;
      inva[row] = inv;
    }
    float* orow = attn + ((size_t)h * S_LEN + (tm << 4) + row) * S_LEN;
    for (int i = 0; i < 32; ++i) {
      int off = (sub + (i << 4)) << 2;
      float4 x = *(const float4*)(srow + off);
      x.x = __expf(x.x - mx) * inv;
      x.y = __expf(x.y - mx) * inv;
      x.z = __expf(x.z - mx) * inv;
      x.w = __expf(x.w - mx) * inv;
      *(float4*)(orow + off) = x;
    }
  }
  __syncthreads();

  // ---- P3: partial ctx tile per wave over a disjoint 256-wide K slice ----
  f32x8 acc[4] = {};
  {
    float mxr = mxa[ln15];
    float invr = inva[ln15];
    const float* arow = sc + ln15 * SROW + ka;
    int kbase = wave << 8;                // 256 K per wave
    for (int kk = 0; kk < 256; kk += 32) {
      int k = kbase + kk;
      const float* ap = arow + k;
      float4 l0 = *(const float4*)(ap);
      float4 l1 = *(const float4*)(ap + 4);
      float4 h0 = *(const float4*)(ap + 16);
      float4 h1 = *(const float4*)(ap + 20);
      bf16x16 a;
      a[0] = (bf16)(__expf(l0.x - mxr) * invr);
      a[1] = (bf16)(__expf(l0.y - mxr) * invr);
      a[2] = (bf16)(__expf(l0.z - mxr) * invr);
      a[3] = (bf16)(__expf(l0.w - mxr) * invr);
      a[4] = (bf16)(__expf(l1.x - mxr) * invr);
      a[5] = (bf16)(__expf(l1.y - mxr) * invr);
      a[6] = (bf16)(__expf(l1.z - mxr) * invr);
      a[7] = (bf16)(__expf(l1.w - mxr) * invr);
      a[8] = (bf16)(__expf(h0.x - mxr) * invr);
      a[9] = (bf16)(__expf(h0.y - mxr) * invr);
      a[10] = (bf16)(__expf(h0.z - mxr) * invr);
      a[11] = (bf16)(__expf(h0.w - mxr) * invr);
      a[12] = (bf16)(__expf(h1.x - mxr) * invr);
      a[13] = (bf16)(__expf(h1.y - mxr) * invr);
      a[14] = (bf16)(__expf(h1.z - mxr) * invr);
      a[15] = (bf16)(__expf(h1.w - mxr) * invr);
      const bf16* pb0 = vt + (size_t)(h * DK_DIM + ln15) * S_LEN + kb + k;
      __builtin_prefetch(pb0 + 512, 0, 1);  // global_prefetch_b8 (L2-resident V^T)
#pragma unroll
      for (int nt = 0; nt < 4; ++nt) {
        acc[nt] = wmma_bf16(a, load_frag_b(pb0 + (size_t)(nt << 4) * S_LEN), acc[nt]);
      }
    }
  }
  __syncthreads();  // all waves done reading sc

  // ---- P4: cross-wave reduction in LDS, store ctx ----
  {
    float* part = sc;  // reuse score buffer as reduction scratch (32 KB)
#pragma unroll
    for (int nt = 0; nt < 4; ++nt)
#pragma unroll
      for (int j = 0; j < 8; ++j)
        part[(wave << 10) + ((mrow + j) << 6) + (nt << 4) + ln15] = acc[nt][j];
    __syncthreads();
    int e = threadIdx.x << 2;             // 256 threads x 4 elems = 16x64 tile
    float4 s = {0.0f, 0.0f, 0.0f, 0.0f};
#pragma unroll
    for (int w = 0; w < 8; ++w) {
      float4 p = *(const float4*)(part + (w << 10) + e);
      s.x += p.x;
      s.y += p.y;
      s.z += p.z;
      s.w += p.w;
    }
    int gm = (tm << 4) + (e >> 6);
    int col = h * DK_DIM + (e & 63);
    *(float4*)(ctxf + (size_t)gm * D_DIM + col) = s;
    bf16x4 b4;
    b4[0] = (bf16)s.x;
    b4[1] = (bf16)s.y;
    b4[2] = (bf16)s.z;
    b4[3] = (bf16)s.w;
    *(bf16x4*)(ctxb + (size_t)gm * D_DIM + col) = b4;
  }
}

// ---- LayerNorm over D=1024 --------------------------------------------------

__global__ void __launch_bounds__(256) layernorm_rows(
    const float* __restrict__ x, const float* __restrict__ gamma,
    const float* __restrict__ beta, float* __restrict__ out) {
  const float* row = x + (size_t)blockIdx.x * D_DIM;
  float* orow = out + (size_t)blockIdx.x * D_DIM;
  __shared__ float red[256];
  int t = threadIdx.x;
  float vals[4];
  float s = 0.0f, s2 = 0.0f;
#pragma unroll
  for (int i = 0; i < 4; ++i) {
    float v = row[t + (i << 8)];
    vals[i] = v;
    s += v;
    s2 += v * v;
  }
  red[t] = s;
  __syncthreads();
  for (int k = 128; k > 0; k >>= 1) {
    if (t < k) red[t] += red[t + k];
    __syncthreads();
  }
  float mean = red[0] * (1.0f / D_DIM);
  __syncthreads();
  red[t] = s2;
  __syncthreads();
  for (int k = 128; k > 0; k >>= 1) {
    if (t < k) red[t] += red[t + k];
    __syncthreads();
  }
  float var = red[0] * (1.0f / D_DIM) - mean * mean;
  float inv = rsqrtf(var + 1e-5f);
#pragma unroll
  for (int i = 0; i < 4; ++i) {
    int c = t + (i << 8);
    orow[c] = (vals[i] - mean) * inv * gamma[c] + beta[c];
  }
}

// ---- Host orchestration -----------------------------------------------------

extern "C" void kernel_launch(void* const* d_in, const int* in_sizes, int n_in,
                              void* d_out, int out_size, void* d_ws, size_t ws_size,
                              hipStream_t stream) {
  const int S = S_LEN, D = D_DIM, H = H_CNT;
  const float* Qf = (const float*)d_in[0];
  const float* Kf = (const float*)d_in[1];
  const float* Vf = (const float*)d_in[2];
  const unsigned char* mask = (const unsigned char*)d_in[3];  // jax bool = 1 byte
  const float* wq = (const float*)d_in[4];
  const float* bq = (const float*)d_in[5];
  const float* wk = (const float*)d_in[6];
  const float* bk = (const float*)d_in[7];
  const float* wv = (const float*)d_in[8];
  const float* bv = (const float*)d_in[9];
  const float* wo = (const float*)d_in[10];
  const float* bo = (const float*)d_in[11];
  const float* gamma = (const float*)d_in[12];
  const float* beta = (const float*)d_in[13];

  // Workspace layout (needs ~44 MB)
  char* ws = (char*)d_ws;
  bf16* Qb = (bf16*)(ws + ((size_t)0 << 20));
  bf16* Kb = (bf16*)(ws + ((size_t)4 << 20));
  bf16* Vb = (bf16*)(ws + ((size_t)8 << 20));
  bf16* wqb = (bf16*)(ws + ((size_t)12 << 20));
  bf16* wkb = (bf16*)(ws + ((size_t)14 << 20));
  bf16* wvb = (bf16*)(ws + ((size_t)16 << 20));
  bf16* wob = (bf16*)(ws + ((size_t)18 << 20));
  bf16* qb = (bf16*)(ws + ((size_t)20 << 20));
  bf16* kb = (bf16*)(ws + ((size_t)24 << 20));
  bf16* vtb = (bf16*)(ws + ((size_t)28 << 20));   // V projection, transposed (D x S)
  bf16* ctxb = (bf16*)(ws + ((size_t)32 << 20));
  float* pre = (float*)(ws + ((size_t)36 << 20));  // pre-LayerNorm fp32 (S x D)

  // d_out layout: out (S*D) ++ attn (H*S*S) ++ ctx2 (S*D)
  float* out_f = (float*)d_out;
  float* attn_f = out_f + (size_t)S * D;
  float* ctx_f = attn_f + (size_t)H * S * S;

  dim3 blk(256);
  auto cvt = [&](const float* src, bf16* dst, int n) {
    cvt_f32_bf16<<<(n / 4 + 255) / 256, blk, 0, stream>>>(src, dst, n);
  };
  cvt(Qf, Qb, S * D);
  cvt(Kf, Kb, S * D);
  cvt(Vf, Vb, S * D);
  cvt(wq, wqb, D * D);
  cvt(wk, wkb, D * D);
  cvt(wv, wvb, D * D);
  cvt(wo, wob, D * D);

  const float inv_sqrt_dk = 0.125f;  // folded into q (safe: q not an output)
  int gemm_blocks = (S / 16) * (D / 16) / 8;  // 1024

  // Projections: q (scaled), k, v (stored transposed for contiguous B-frags).
  wmma_gemm_bias<<<gemm_blocks, blk, 0, stream>>>(Qb, wqb, bq, nullptr, inv_sqrt_dk,
                                                  qb, D, 1, nullptr, S, D, D);
  wmma_gemm_bias<<<gemm_blocks, blk, 0, stream>>>(Kb, wkb, bk, nullptr, 1.0f,
                                                  kb, D, 1, nullptr, S, D, D);
  wmma_gemm_bias<<<gemm_blocks, blk, 0, stream>>>(Vb, wvb, bv, nullptr, 1.0f,
                                                  vtb, 1, S, nullptr, S, D, D);

  // Fused scores + mask + softmax + attn@V. attn hits HBM exactly once.
  size_t smem = (size_t)LDS_QSTAGE_OFF + 16 * DK_DIM * sizeof(bf16);  // ~130.4 KB
  wmma_attn_fused<<<H * (S / 16), blk, smem, stream>>>(qb, kb, mask, vtb,
                                                       attn_f, ctx_f, ctxb);

  // out-projection + bias + residual(Q) -> pre-LN scratch, then LayerNorm.
  wmma_gemm_bias<<<gemm_blocks, blk, 0, stream>>>(ctxb, wob, bo, Qf, 1.0f,
                                                  nullptr, 0, 0, pre, S, D, D);
  layernorm_rows<<<S, blk, 0, stream>>>(pre, gamma, beta, out_f);
}
